// CM_FEAT_52913997087039
// MI455X (gfx1250) — matrix operand order repaired and checked by reading, not compiled, not run on hardware
//
#include <hip/hip_runtime.h>
#include <hip/hip_bf16.h>
#include <math.h>
#include <stdint.h>

typedef __attribute__((ext_vector_type(16))) _Float16 v16h;
typedef __attribute__((ext_vector_type(8)))  float    v8f;
typedef __attribute__((ext_vector_type(2)))  _Float16 h2v;

#define DEPTH_ 6

// ---------------------------------------------------------------------------
// WMMA fragment loaders per CDNA5 ISA 7.12.2 (wave32, 16-bit A/B, f32 C/D)
// ---------------------------------------------------------------------------
__device__ __forceinline__ v16h load_A_frag(const _Float16* A, int lda,
                                            int m0, int k0, int hf, int lr) {
  v16h a;
  const _Float16* base = A + (m0 + lr) * lda + k0;
#pragma unroll
  for (int v = 0; v < 8; ++v) {
    int K = 2 * v + 8 * hf + ((v >= 4) ? 8 : 0);
    h2v p = *(const h2v*)(base + K);
    a[2 * v] = p[0];
    a[2 * v + 1] = p[1];
  }
  return a;
}

// B: 32x16 (KxN) with B[k][n] = W[n][k], W row-major (Nout x Ktot), ld ldw.
__device__ __forceinline__ v16h load_B_frag(const _Float16* W, int ldw,
                                            int k0, int n0, int hf, int lr) {
  v16h b;
  const _Float16* base = W + (n0 + lr) * ldw + k0;
#pragma unroll
  for (int r = 0; r < 8; ++r) {
    int K = 2 * r + 8 * hf + ((r >= 4) ? 8 : 0);
    h2v p = *(const h2v*)(base + K);
    b[2 * r] = p[0];
    b[2 * r + 1] = p[1];
  }
  return b;
}

// Y(MxN f32, LDS) = A(f16 LDS) @ W(f16 LDS, Nout x K row-major).T [+bias][+Y]
__device__ __forceinline__ void wave_gemm(const _Float16* A, int lda,
                                          const _Float16* W, int ldw,
                                          float* D, int ldd,
                                          const float* bias, bool addD,
                                          int Mt, int Nt, int Kt,
                                          int wave, int nwaves, int lane) {
  const int hf = lane >> 4, lr = lane & 15;
  const int ntiles = Mt * Nt;
  for (int t = wave; t < ntiles; t += nwaves) {
    const int mt = t / Nt, nt = t % Nt;
    v8f c = {0.f, 0.f, 0.f, 0.f, 0.f, 0.f, 0.f, 0.f};
    for (int kt = 0; kt < Kt; ++kt) {
      v16h a = load_A_frag(A, lda, mt * 16, kt * 32, hf, lr);
      v16h b = load_B_frag(W, ldw, kt * 32, nt * 16, hf, lr);
      c = __builtin_amdgcn_wmma_f32_16x16x32_f16(false, a, false, b,
                                                 (short)0, c, false, false);
    }
    const int col = nt * 16 + lr;
    const float bv = bias ? bias[col] : 0.f;
#pragma unroll
    for (int r = 0; r < 8; ++r) {
      const int row = mt * 16 + r + 8 * hf;
      float acc = c[r] + bv;
      if (addD) acc += D[row * ldd + col];
      D[row * ldd + col] = acc;
    }
  }
}

// ---------------------------------------------------------------------------
// Async global -> LDS staging (CDNA5 ASYNCcnt path).
// Each active lane moves 16 bytes per issued op; wait via s_wait_asynccnt.
// ---------------------------------------------------------------------------
__device__ __forceinline__ void async_issue_copy16(const void* g, void* l,
                                                   int bytes, int tid) {
  unsigned lbase = (unsigned)(uintptr_t)l;  // flat LDS addr, low 32 = offset
  const char* gb = (const char*)g;
  for (int off = tid * 16; off < bytes; off += 256 * 16) {
    unsigned laddr = lbase + (unsigned)off;
    const void* ga = gb + off;
    asm volatile("global_load_async_to_lds_b128 %0, %1, off"
                 :: "v"(laddr), "v"(ga)
                 : "memory");
  }
}

__device__ __forceinline__ void async_wait_all() {
  asm volatile("s_wait_asynccnt 0x0" ::: "memory");
}

__device__ __forceinline__ float gelu_exact(float x) {
  return 0.5f * x * (1.f + erff(x * 0.70710678118654752f));
}

// ---------------------------------------------------------------------------
// Kernel 1: frontend -> rotated signal -> per-block min/max
// ---------------------------------------------------------------------------
__global__ __launch_bounds__(256) void front_minmax_kernel(
    const float* __restrict__ x, const float* __restrict__ fc1_w,
    const float* __restrict__ fc1_b, const float* __restrict__ fc2_w,
    const float* __restrict__ fc2_b, float* __restrict__ bmin,
    float* __restrict__ bmax) {
  __shared__ float xs[2048];
  __shared__ float hbuf[80];
  __shared__ float wf[2];
  __shared__ float rmin[256], rmax[256];
  const int b = blockIdx.x, tid = threadIdx.x;
  for (int i = tid; i < 2048; i += 256) xs[i] = x[b * 2048 + i];
  __syncthreads();
  if (tid < 80) {
    float acc = fc1_b[tid];
    const float* wr = fc1_w + tid * 2048;
    for (int d = 0; d < 2048; ++d) acc += xs[d] * wr[d];
    hbuf[tid] = tanhf(acc);
  }
  __syncthreads();
  if (tid < 2) {
    float acc = fc2_b[tid];
    const float* wr = fc2_w + tid * 80;
    for (int k = 0; k < 80; ++k) acc += hbuf[k] * wr[k];
    wf[tid] = acc;
  }
  __syncthreads();
  const float w0 = wf[0], phi = wf[1];
  float mn = 3.4e38f, mx = -3.4e38f;
  for (int n = tid; n < 1024; n += 256) {
    const float ang = w0 * (float)n + phi;
    const float cs = cosf(ang), sn = sinf(ang);
    const float R = xs[n], I = xs[1024 + n];
    const float a0 = R * cs + I * sn;
    const float a1 = I * cs - R * sn;
    mn = fminf(mn, fminf(a0, a1));
    mx = fmaxf(mx, fmaxf(a0, a1));
  }
  rmin[tid] = mn;
  rmax[tid] = mx;
  __syncthreads();
  for (int s = 128; s > 0; s >>= 1) {
    if (tid < s) {
      rmin[tid] = fminf(rmin[tid], rmin[tid + s]);
      rmax[tid] = fmaxf(rmax[tid], rmax[tid + s]);
    }
    __syncthreads();
  }
  if (tid == 0) {
    bmin[b] = rmin[0];
    bmax[b] = rmax[0];
  }
}

// ---------------------------------------------------------------------------
// Kernel 2: reduce 1024 block min/max -> global
// ---------------------------------------------------------------------------
__global__ __launch_bounds__(256) void minmax_reduce_kernel(
    const float* __restrict__ bmin, const float* __restrict__ bmax,
    float* __restrict__ gmm) {
  __shared__ float rmin[256], rmax[256];
  const int tid = threadIdx.x;
  float mn = 3.4e38f, mx = -3.4e38f;
  for (int i = tid; i < 1024; i += 256) {
    mn = fminf(mn, bmin[i]);
    mx = fmaxf(mx, bmax[i]);
  }
  rmin[tid] = mn;
  rmax[tid] = mx;
  __syncthreads();
  for (int s = 128; s > 0; s >>= 1) {
    if (tid < s) {
      rmin[tid] = fminf(rmin[tid], rmin[tid + s]);
      rmax[tid] = fmaxf(rmax[tid], rmax[tid + s]);
    }
    __syncthreads();
  }
  if (tid == 0) {
    gmm[0] = rmin[0];
    gmm[1] = rmax[0];
  }
}

// ---------------------------------------------------------------------------
// Kernel 3: weights -> f16 in workspace (We transposed so B-loader fits)
// ---------------------------------------------------------------------------
__global__ __launch_bounds__(256) void prep_weights_kernel(
    const float* __restrict__ qkv_w, const float* __restrict__ out_w,
    const float* __restrict__ ff1_w, const float* __restrict__ ff2_w,
    const float* __restrict__ ff3_w, const float* __restrict__ We,
    _Float16* __restrict__ qkv16, _Float16* __restrict__ out16,
    _Float16* __restrict__ ff1h, _Float16* __restrict__ ff2h,
    _Float16* __restrict__ ff3h, _Float16* __restrict__ WeT) {
  const int i = blockIdx.x * blockDim.x + threadIdx.x;
  const int s0 = 73728, s1 = 98304, s2 = 147456, s3 = 196608, s4 = 245760,
            s5 = 249856;
  if (i < s0) {
    qkv16[i] = (_Float16)qkv_w[i];
  } else if (i < s1) {
    const int j = i - s0;
    out16[j] = (_Float16)out_w[j];
  } else if (i < s2) {
    const int j = i - s1;
    ff1h[j] = (_Float16)ff1_w[j];
  } else if (i < s3) {
    const int j = i - s2;
    ff2h[j] = (_Float16)ff2_w[j];
  } else if (i < s4) {
    const int j = i - s3;
    ff3h[j] = (_Float16)ff3_w[j];
  } else if (i < s5) {
    const int j = i - s4;
    const int e = j >> 6, d = j & 63;
    WeT[e * 64 + d] = (_Float16)We[d * 64 + e];
  }
}

// ---------------------------------------------------------------------------
// Kernel 4: fused per-sample transformer
// ---------------------------------------------------------------------------
__global__ __launch_bounds__(256) void vit_main_kernel(
    const float* __restrict__ x, const float* __restrict__ fc1_w,
    const float* __restrict__ fc1_b, const float* __restrict__ fc2_w,
    const float* __restrict__ fc2_b, const float* __restrict__ pos_emb,
    const float* __restrict__ cls_token, const float* __restrict__ ln1_g,
    const float* __restrict__ ln1_b, const float* __restrict__ th1_w,
    const float* __restrict__ th2_w, const float* __restrict__ out_b,
    const float* __restrict__ ln2_g, const float* __restrict__ ln2_b,
    const float* __restrict__ ff1_b, const float* __restrict__ ff2_b,
    const float* __restrict__ ffln_g, const float* __restrict__ ffln_b,
    const float* __restrict__ ff3_b, const float* __restrict__ fln_g,
    const float* __restrict__ fln_b, const float* __restrict__ head_w,
    const float* __restrict__ head_b, const _Float16* __restrict__ qkv16,
    const _Float16* __restrict__ out16, const _Float16* __restrict__ ff1h,
    const _Float16* __restrict__ ff2h, const _Float16* __restrict__ ff3h,
    const _Float16* __restrict__ WeT, const float* __restrict__ gmm,
    float* __restrict__ out) {
  __shared__ float xt[64][64];                     // activations (16 KB)
  __shared__ _Float16 a16[64][128];                // WMMA A staging (16 KB)
  __shared__ float fbuf[64][192];                  // qkv / ff1 / front (48 KB)
  __shared__ float sc[16][8][64];                  // scores / ff2 out (32 KB)
  __shared__ alignas(16) _Float16 wst[12288];      // weight stage A (24 KB)
  __shared__ alignas(16) _Float16 wst2[8192];      // weight stage B (16 KB)
  __shared__ float th1s[64], th2s[64];
  __shared__ float red[2];

  const int b = blockIdx.x, tid = threadIdx.x;
  const int wave = tid >> 5, lane = tid & 31;
  float* fl = &fbuf[0][0];
  float* pe = (float*)wst2;  // pos_emb staged here for the embed phase

  // Kick off async staging of We^T and pos_emb while the frontend runs.
  async_issue_copy16(WeT, wst, 8192, tid);
  async_issue_copy16(pos_emb, wst2, 16384, tid);

  // ---- frontend recompute ----
  for (int i = tid; i < 2048; i += 256) fl[i] = x[b * 2048 + i];
  __syncthreads();
  if (tid < 80) {
    float acc = fc1_b[tid];
    const float* wr = fc1_w + tid * 2048;
    for (int d = 0; d < 2048; ++d) acc += fl[d] * wr[d];
    fl[2048 + tid] = tanhf(acc);
  }
  __syncthreads();
  if (tid < 2) {
    float acc = fc2_b[tid];
    const float* wr = fc2_w + tid * 80;
    for (int k = 0; k < 80; ++k) acc += fl[2048 + k] * wr[k];
    fl[2130 + tid] = acc;
  }
  __syncthreads();
  const float w0 = fl[2130], phi = fl[2131];
  const float gmin = gmm[0];
  const float ginv = 1.f / (gmm[1] - gmin);

  // ---- tokenize directly into f16 A buffer (row 63 zero-padded) ----
  for (int e = tid; e < 4096; e += 256) {
    const int m = e >> 6, c = e & 63;
    float tval = 0.f;
    if (m < 63) {
      const int ch = c >> 5, j = c & 31, n = m * 16 + j;
      const float ang = w0 * (float)n + phi;
      const float cs = cosf(ang), sn = sinf(ang);
      const float R = fl[n], I = fl[1024 + n];
      const float v = (ch == 0) ? (R * cs + I * sn) : (I * cs - R * sn);
      tval = (v - gmin) * ginv;
    }
    a16[m][c] = (_Float16)tval;
  }
  async_wait_all();
  __syncthreads();

  // ---- token embedding: xt[1+m] = tok @ We + pos_emb (WMMA, LDS weights) ----
  {
    const int hf = lane >> 4, lr = lane & 15;
    for (int t = wave; t < 16; t += 8) {
      const int mt = t >> 2, nt = t & 3;
      v8f c = {0.f, 0.f, 0.f, 0.f, 0.f, 0.f, 0.f, 0.f};
      for (int kt = 0; kt < 2; ++kt) {
        v16h a = load_A_frag(&a16[0][0], 128, mt * 16, kt * 32, hf, lr);
        v16h bf = load_B_frag(wst, 64, kt * 32, nt * 16, hf, lr);
        c = __builtin_amdgcn_wmma_f32_16x16x32_f16(false, a, false, bf,
                                                   (short)0, c, false, false);
      }
      const int col = nt * 16 + lr;
#pragma unroll
      for (int r = 0; r < 8; ++r) {
        const int m = mt * 16 + r + 8 * hf;
        if (m < 63) xt[m + 1][col] = c[r] + pe[(m + 1) * 64 + col];
      }
    }
  }
  if (tid < 64) xt[0][tid] = cls_token[tid] + pe[tid];
  __syncthreads();

  // ---- 6 transformer layers ----
  for (int l = 0; l < DEPTH_; ++l) {
    // stage qkv weights while LN1 runs
    async_issue_copy16(qkv16 + l * 12288, wst, 24576, tid);
    if (tid < 64) {
      float mean = 0.f;
      for (int c = 0; c < 64; ++c) mean += xt[tid][c];
      mean *= (1.f / 64.f);
      float var = 0.f;
      for (int c = 0; c < 64; ++c) {
        const float dv = xt[tid][c] - mean;
        var += dv * dv;
      }
      var *= (1.f / 64.f);
      const float rs = rsqrtf(var + 1e-5f);
      for (int c = 0; c < 64; ++c)
        a16[tid][c] = (_Float16)((xt[tid][c] - mean) * rs * ln1_g[l * 64 + c] +
                                 ln1_b[l * 64 + c]);
    } else if (tid < 128) {
      th1s[tid - 64] = th1_w[l * 64 + (tid - 64)];
    } else if (tid < 192) {
      th2s[tid - 128] = th2_w[l * 64 + (tid - 128)];
    }
    async_wait_all();
    __syncthreads();

    // qkv = LN1(x) @ qkv_w.T  (64x192, WMMA, weights in LDS)
    wave_gemm(&a16[0][0], 128, wst, 64, &fbuf[0][0], 192, nullptr, false, 4,
              12, 2, wave, 8, lane);
    __syncthreads();

    // stage out-proj weights while attention runs
    async_issue_copy16(out16 + l * 4096, wst, 8192, tid);

    // talking-heads attention, 16-row blocks (VALU; head_dim=8)
    for (int rb = 0; rb < 4; ++rb) {
      for (int it = tid; it < 1024; it += 256) {
        const int il = it >> 6, j = it & 63, i = rb * 16 + il;
        float d[8];
#pragma unroll
        for (int h = 0; h < 8; ++h) {
          float acc = 0.f;
#pragma unroll
          for (int dd = 0; dd < 8; ++dd)
            acc += fbuf[i][h * 8 + dd] * fbuf[j][64 + h * 8 + dd];
          d[h] = acc * 0.35355339059327373f;  // DH^-0.5
        }
#pragma unroll
        for (int g = 0; g < 8; ++g) {
          float m = 0.f;
#pragma unroll
          for (int h = 0; h < 8; ++h) m += th1s[g * 8 + h] * d[h];
          sc[il][g][j] = m;
        }
      }
      __syncthreads();
      if (tid < 128) {
        const int il = tid >> 3, g = tid & 7;
        float* r = sc[il][g];
        float mx = -3.4e38f;
        for (int j = 0; j < 64; ++j) mx = fmaxf(mx, r[j]);
        float sum = 0.f;
        for (int j = 0; j < 64; ++j) {
          const float e = expf(r[j] - mx);
          r[j] = e;
          sum += e;
        }
        const float is = 1.f / sum;
        for (int j = 0; j < 64; ++j) r[j] *= is;
      }
      __syncthreads();
      for (int it = tid; it < 1024; it += 256) {
        const int il = it >> 6, j = it & 63;
        float s[8];
#pragma unroll
        for (int h = 0; h < 8; ++h) s[h] = sc[il][h][j];
#pragma unroll
        for (int g = 0; g < 8; ++g) {
          float m = 0.f;
#pragma unroll
          for (int h = 0; h < 8; ++h) m += th2s[g * 8 + h] * s[h];
          sc[il][g][j] = m;
        }
      }
      __syncthreads();
      for (int it = tid; it < 1024; it += 256) {
        const int il = it >> 6, o = it & 63, g = o >> 3;
        float acc = 0.f;
        for (int j = 0; j < 64; ++j) acc += sc[il][g][j] * fbuf[j][128 + o];
        a16[rb * 16 + il][o] = (_Float16)acc;
      }
      __syncthreads();
    }
    async_wait_all();
    __syncthreads();

    // out-proj + bias + residual into xt (WMMA, weights in LDS)
    wave_gemm(&a16[0][0], 128, wst, 64, &xt[0][0], 64, out_b + l * 64, true, 4,
              4, 2, wave, 8, lane);
    __syncthreads();

    // stage ff1/ff2 weights while LN2 runs
    async_issue_copy16(ff1h + l * 8192, wst, 16384, tid);
    async_issue_copy16(ff2h + l * 8192, wst2, 16384, tid);
    if (tid < 64) {
      float mean = 0.f;
      for (int c = 0; c < 64; ++c) mean += xt[tid][c];
      mean *= (1.f / 64.f);
      float var = 0.f;
      for (int c = 0; c < 64; ++c) {
        const float dv = xt[tid][c] - mean;
        var += dv * dv;
      }
      var *= (1.f / 64.f);
      const float rs = rsqrtf(var + 1e-5f);
      for (int c = 0; c < 64; ++c)
        a16[tid][c] = (_Float16)((xt[tid][c] - mean) * rs * ln2_g[l * 64 + c] +
                                 ln2_b[l * 64 + c]);
    }
    async_wait_all();
    __syncthreads();

    // x1 -> fbuf[:,0:128], x2 -> sc (viewed 64x128)  (WMMA)
    wave_gemm(&a16[0][0], 128, wst, 64, &fbuf[0][0], 192, ff1_b + l * 128,
              false, 4, 8, 2, wave, 8, lane);
    wave_gemm(&a16[0][0], 128, wst2, 64, &sc[0][0][0], 128, ff2_b + l * 128,
              false, 4, 8, 2, wave, 8, lane);
    __syncthreads();

    // stage ff3 weights while the gated GELU + inner LN runs
    async_issue_copy16(ff3h + l * 8192, wst, 16384, tid);
    if (tid < 64) {
      float* x2 = &sc[0][0][0] + tid * 128;
      const float* x1 = &fbuf[tid][0];
      float mean = 0.f;
      for (int c = 0; c < 128; ++c) {
        const float v1 = x1[c], v2 = x2[c];
        const float f = v1 * gelu_exact(v2) + v2 * gelu_exact(v1);
        x2[c] = f;
        mean += f;
      }
      mean *= (1.f / 128.f);
      float var = 0.f;
      for (int c = 0; c < 128; ++c) {
        const float dv = x2[c] - mean;
        var += dv * dv;
      }
      var *= (1.f / 128.f);
      const float rs = rsqrtf(var + 1e-5f);
      for (int c = 0; c < 128; ++c)
        a16[tid][c] = (_Float16)((x2[c] - mean) * rs * ffln_g[l * 128 + c] +
                                 ffln_b[l * 128 + c]);
    }
    async_wait_all();
    __syncthreads();

    // ff3 + bias + residual into xt (WMMA, K=128)
    wave_gemm(&a16[0][0], 128, wst, 128, &xt[0][0], 64, ff3_b + l * 64, true,
              4, 4, 4, wave, 8, lane);
    __syncthreads();
  }

  // ---- final LN on cls + head ----
  if (tid == 0) {
    float mean = 0.f;
    for (int c = 0; c < 64; ++c) mean += xt[0][c];
    mean *= (1.f / 64.f);
    float var = 0.f;
    for (int c = 0; c < 64; ++c) {
      const float dv = xt[0][c] - mean;
      var += dv * dv;
    }
    var *= (1.f / 64.f);
    red[0] = mean;
    red[1] = rsqrtf(var + 1e-5f);
  }
  __syncthreads();
  if (tid < 10) {
    const float mean = red[0], rs = red[1];
    float acc = head_b[tid];
    for (int d = 0; d < 64; ++d)
      acc += ((xt[0][d] - mean) * rs * fln_g[d] + fln_b[d]) *
             head_w[tid * 64 + d];
    out[b * 10 + tid] = acc;
  }
}

// ---------------------------------------------------------------------------
extern "C" void kernel_launch(void* const* d_in, const int* in_sizes, int n_in,
                              void* d_out, int out_size, void* d_ws,
                              size_t ws_size, hipStream_t stream) {
  (void)in_sizes; (void)n_in; (void)out_size; (void)ws_size;
  const float* x        = (const float*)d_in[0];
  const float* fc1_w    = (const float*)d_in[1];
  const float* fc1_b    = (const float*)d_in[2];
  const float* fc2_w    = (const float*)d_in[3];
  const float* fc2_b    = (const float*)d_in[4];
  const float* pos_emb  = (const float*)d_in[5];
  const float* cls_tok  = (const float*)d_in[6];
  const float* We       = (const float*)d_in[7];
  const float* ln1_g    = (const float*)d_in[8];
  const float* ln1_b    = (const float*)d_in[9];
  const float* qkv_w    = (const float*)d_in[10];
  const float* th1_w    = (const float*)d_in[11];
  const float* th2_w    = (const float*)d_in[12];
  const float* out_w    = (const float*)d_in[13];
  const float* out_b    = (const float*)d_in[14];
  const float* ln2_g    = (const float*)d_in[15];
  const float* ln2_b    = (const float*)d_in[16];
  const float* ff1_w    = (const float*)d_in[17];
  const float* ff1_b    = (const float*)d_in[18];
  const float* ff2_w    = (const float*)d_in[19];
  const float* ff2_b    = (const float*)d_in[20];
  const float* ffln_g   = (const float*)d_in[21];
  const float* ffln_b   = (const float*)d_in[22];
  const float* ff3_w    = (const float*)d_in[23];
  const float* ff3_b    = (const float*)d_in[24];
  const float* fln_g    = (const float*)d_in[25];
  const float* fln_b    = (const float*)d_in[26];
  const float* head_w   = (const float*)d_in[27];
  const float* head_b   = (const float*)d_in[28];
  float* out = (float*)d_out;

  // workspace layout
  float* bmin = (float*)d_ws;              // 1024 f32
  float* bmax = bmin + 1024;               // 1024 f32
  float* gmm  = bmax + 1024;               // 2 f32
  _Float16* qkv16 = (_Float16*)((char*)d_ws + 8448);  // 256B-aligned halves
  _Float16* out16 = qkv16 + 6 * 192 * 64;
  _Float16* ff1h  = out16 + 6 * 64 * 64;
  _Float16* ff2h  = ff1h + 6 * 128 * 64;
  _Float16* ff3h  = ff2h + 6 * 128 * 64;
  _Float16* WeT   = ff3h + 6 * 64 * 128;

  front_minmax_kernel<<<1024, 256, 0, stream>>>(x, fc1_w, fc1_b, fc2_w, fc2_b,
                                                bmin, bmax);
  minmax_reduce_kernel<<<1, 256, 0, stream>>>(bmin, bmax, gmm);
  prep_weights_kernel<<<976, 256, 0, stream>>>(qkv_w, out_w, ff1_w, ff2_w,
                                               ff3_w, We, qkv16, out16, ff1h,
                                               ff2h, ff3h, WeT);
  vit_main_kernel<<<1024, 256, 0, stream>>>(
      x, fc1_w, fc1_b, fc2_w, fc2_b, pos_emb, cls_tok, ln1_g, ln1_b, th1_w,
      th2_w, out_b, ln2_g, ln2_b, ff1_b, ff2_b, ffln_g, ffln_b, ff3_b, fln_g,
      fln_b, head_w, head_b, qkv16, out16, ff1h, ff2h, ff3h, WeT, gmm, out);
}
